// PanModel_19490561590285
// MI455X (gfx1250) — compile-verified, matrix operand order checked by reading, not run on hardware
//
#include <hip/hip_runtime.h>
#include <math.h>

typedef __attribute__((ext_vector_type(16))) _Float16 v16h;
typedef __attribute__((ext_vector_type(8)))  float    v8f;
typedef _Float16 half_t;

#define NB      8192
#define EMB     32
#define SEQ     15
#define MHC     74
#define MHC_PAD 80
#define W16_TOTAL 67584

#if __has_builtin(__builtin_amdgcn_global_load_async_to_lds_b128)
#define HAVE_ASYNC_LDS 1
typedef int v4i_vec __attribute__((vector_size(16)));
typedef __attribute__((address_space(1))) const char gchar_c;
typedef __attribute__((address_space(3))) char lchar;
typedef __attribute__((address_space(1))) v4i_vec gvec4;
typedef __attribute__((address_space(3))) v4i_vec lvec4;
#endif

// ---- wave-local synchronization -------------------------------------------
// Each wave owns a private LDS slice; cross-lane LDS dependencies only need
// the wave's own DS ops to complete (DScnt) plus a compiler reorder fence.
// This avoids s_barrier_signal/s_barrier_wait entirely.
__device__ __forceinline__ void wave_sync() {
  __builtin_amdgcn_wave_barrier();
  asm volatile("s_wait_dscnt 0x0" ::: "memory");
  __builtin_amdgcn_wave_barrier();
}

__device__ __forceinline__ void async_copy_wait() {
#if defined(HAVE_ASYNC_LDS)
#if __has_builtin(__builtin_amdgcn_s_wait_asynccnt)
  __builtin_amdgcn_s_wait_asynccnt(0);
  asm volatile("" ::: "memory");
#else
  asm volatile("s_wait_asynccnt 0x0" ::: "memory");
#endif
#endif
}

// ---------------- WMMA fragment helpers (CDNA5 ISA layouts, wave32) --------

__device__ __forceinline__ int frag_k(int h, int lane) {
  // 16-bit A/B fragment: half-index h -> K index
  return ((h & 8) << 1) + ((lane & 16) >> 1) + (h & 7);
}

__device__ __forceinline__ v16h frag_a_f32(const float* A, int lda, int lane) {
  int m = lane & 15;
  v16h a;
#pragma unroll
  for (int h = 0; h < 16; ++h) a[h] = (_Float16)A[m * lda + frag_k(h, lane)];
  return a;
}

__device__ __forceinline__ v16h frag_a_f16(const half_t* A, int lda, int lane) {
  int m = lane & 15;
  v16h a;
#pragma unroll
  for (int h = 0; h < 16; ++h) a[h] = A[m * lda + frag_k(h, lane)];
  return a;
}

// B fragment for computing X @ W^T : B[k][n] = W[n][k]; W row-major [out][in]
__device__ __forceinline__ v16h frag_b_f16(const half_t* W, int ldw, int lane) {
  int n = lane & 15;
  v16h b;
#pragma unroll
  for (int h = 0; h < 16; ++h) b[h] = W[n * ldw + frag_k(h, lane)];
  return b;
}

__device__ __forceinline__ v8f frag_c_bias(const float* bias, int lane) {
  float bv = bias[lane & 15];
  v8f c;
#pragma unroll
  for (int r = 0; r < 8; ++r) c[r] = bv;
  return c;
}

__device__ __forceinline__ void store_c_f32(float* D, int ldd, v8f c, int lane) {
  int n = lane & 15, mb = (lane & 16) >> 1;
#pragma unroll
  for (int r = 0; r < 8; ++r) D[(mb + r) * ldd + n] = c[r];
}

__device__ __forceinline__ void store_c_add(float* D, int ldd, v8f c, int lane) {
  int n = lane & 15, mb = (lane & 16) >> 1;
#pragma unroll
  for (int r = 0; r < 8; ++r) D[(mb + r) * ldd + n] += c[r];
}

__device__ __forceinline__ void store_c_f16(half_t* D, int ldd, v8f c, int lane) {
  int n = lane & 15, mb = (lane & 16) >> 1;
#pragma unroll
  for (int r = 0; r < 8; ++r) D[(mb + r) * ldd + n] = (_Float16)c[r];
}

__device__ __forceinline__ v8f wmma32(v16h a, v16h b, v8f c) {
  return __builtin_amdgcn_wmma_f32_16x16x32_f16(false, a, false, b, (short)0, c, false, false);
}

// ---------------- per-wave LDS state ----------------

struct WaveLds {
  float  cmhc[MHC_PAD * EMB];   // 10240 B  (this sample's context, f32)
  half_t khf[MHC_PAD * EMB];    //  5120 B  (keys after double projection, f16)
  half_t vhf[MHC_PAD * EMB];    //  5120 B
  float  x[16 * EMB];           //  2048 B  (residual stream, rows 0..14 valid)
  float  xn[16 * EMB];          //  2048 B  (LayerNorm output)
  union {
    struct { float q[16 * EMB]; float tmp[16 * EMB]; } a;  // 4096 B
    half_t hbuf[16 * 128];                                  // 4096 B (FFN hidden, f16)
  } u;
};  // 28672 B per wave, 2 waves/block = 57344 B

// ---------------- building blocks ----------------

__device__ __forceinline__ void layer_norm(const float* x, float* xn,
                                           const float* g, const float* b, int lane) {
  for (int row = 0; row < SEQ; ++row) {
    float v = x[row * EMB + lane];
    float s = v, s2 = v * v;
#pragma unroll
    for (int off = 16; off; off >>= 1) {
      s  += __shfl_xor(s,  off, 32);
      s2 += __shfl_xor(s2, off, 32);
    }
    float m   = s * (1.0f / 32.0f);
    float var = s2 * (1.0f / 32.0f) - m * m;
    xn[row * EMB + lane] = (v - m) * rsqrtf(var + 1e-5f) * g[lane] + b[lane];
  }
  xn[SEQ * EMB + lane] = 0.0f;  // zero the padding row
}

// dst(16x32 f32) = src(16x32 f32) @ W^T + bias   (2 wmma)
__device__ __forceinline__ void proj32_f32(float* dst, const float* src,
                                           const half_t* W, const float* bias, int lane) {
  v16h a = frag_a_f32(src, EMB, lane);
#pragma unroll
  for (int nt = 0; nt < 2; ++nt) {
    v16h b = frag_b_f16(W + nt * 16 * EMB, EMB, lane);
    v8f  c = frag_c_bias(bias + nt * 16, lane);
    c = wmma32(a, b, c);
    store_c_f32(dst + nt * 16, EMB, c, lane);
  }
}

__device__ __forceinline__ void proj32_f16(half_t* dst, const float* src,
                                           const half_t* W, const float* bias, int lane) {
  v16h a = frag_a_f32(src, EMB, lane);
#pragma unroll
  for (int nt = 0; nt < 2; ++nt) {
    v16h b = frag_b_f16(W + nt * 16 * EMB, EMB, lane);
    v8f  c = frag_c_bias(bias + nt * 16, lane);
    c = wmma32(a, b, c);
    store_c_f16(dst + nt * 16, EMB, c, lane);
  }
}

// dst(16x32 f32) += src @ W^T + bias
__device__ __forceinline__ void proj32_add(float* dst, const float* src,
                                           const half_t* W, const float* bias, int lane) {
  v16h a = frag_a_f32(src, EMB, lane);
#pragma unroll
  for (int nt = 0; nt < 2; ++nt) {
    v16h b = frag_b_f16(W + nt * 16 * EMB, EMB, lane);
    v8f  c = frag_c_bias(bias + nt * 16, lane);
    c = wmma32(a, b, c);
    store_c_add(dst + nt * 16, EMB, c, lane);
  }
}

// out(80x32 f16) = ((cmhc @ W1^T + b1) @ W2^T + b2)  -- two WMMA passes, stage2 in place
__device__ __forceinline__ void cross_kv(half_t* out, const float* cm,
                                         const half_t* W1, const float* b1,
                                         const half_t* W2, const float* b2, int lane) {
#pragma unroll
  for (int mt = 0; mt < 5; ++mt) {
    v16h a = frag_a_f32(cm + mt * 16 * EMB, EMB, lane);
#pragma unroll
    for (int nt = 0; nt < 2; ++nt) {
      v16h b = frag_b_f16(W1 + nt * 16 * EMB, EMB, lane);
      v8f  c = frag_c_bias(b1 + nt * 16, lane);
      c = wmma32(a, b, c);
      store_c_f16(out + mt * 16 * EMB + nt * 16, EMB, c, lane);
    }
  }
  wave_sync();
#pragma unroll
  for (int mt = 0; mt < 5; ++mt) {
    v16h a  = frag_a_f16(out + mt * 16 * EMB, EMB, lane);  // read full tile first
    v8f  c0 = frag_c_bias(b2, lane);
    v8f  c1 = frag_c_bias(b2 + 16, lane);
    c0 = wmma32(a, frag_b_f16(W2, EMB, lane), c0);
    c1 = wmma32(a, frag_b_f16(W2 + 16 * EMB, EMB, lane), c1);
    store_c_f16(out + mt * 16 * EMB, EMB, c0, lane);
    store_c_f16(out + mt * 16 * EMB + 16, EMB, c1, lane);
  }
}

// online-softmax attention; lane = head(2b) x rowoff(3b); Dh = 8
__device__ __forceinline__ void attention(const float* q, const half_t* kh, const half_t* vh,
                                          int Sk, float* o, int lane) {
  const int hd   = lane >> 3;
  const int roff = lane & 7;
  const float scale = 0.35355339059327373f;  // 1/sqrt(8)
#pragma unroll
  for (int chunk = 0; chunk < 2; ++chunk) {
    int  row   = chunk * 8 + roff;
    bool valid = row < SEQ;
    float qreg[8];
#pragma unroll
    for (int d = 0; d < 8; ++d) qreg[d] = valid ? q[row * EMB + hd * 8 + d] : 0.0f;
    float mmax = -1e30f, lsum = 0.0f;
    float acc[8];
#pragma unroll
    for (int d = 0; d < 8; ++d) acc[d] = 0.0f;
    for (int j = 0; j < Sk; ++j) {
      float s = 0.0f;
#pragma unroll
      for (int d = 0; d < 8; ++d) s += qreg[d] * (float)kh[j * EMB + hd * 8 + d];
      s *= scale;
      float nm   = fmaxf(mmax, s);
      float corr = __expf(mmax - nm);
      float p    = __expf(s - nm);
      lsum = lsum * corr + p;
#pragma unroll
      for (int d = 0; d < 8; ++d)
        acc[d] = acc[d] * corr + p * (float)vh[j * EMB + hd * 8 + d];
      mmax = nm;
    }
    float inv = 1.0f / lsum;
#pragma unroll
    for (int d = 0; d < 8; ++d)
      o[row * EMB + hd * 8 + d] = valid ? acc[d] * inv : 0.0f;
  }
}

// FFN: x += gelu(xn @ fcW^T + fcB) @ projW^T + projB
__device__ __forceinline__ void ffn(WaveLds& L, const half_t* fcW, const float* fcB,
                                    const half_t* projW, const float* projB, int lane) {
  v16h a = frag_a_f32(L.xn, EMB, lane);
#pragma unroll
  for (int nt = 0; nt < 8; ++nt) {
    v16h b = frag_b_f16(fcW + nt * 16 * EMB, EMB, lane);
    v8f  c = frag_c_bias(fcB + nt * 16, lane);
    c = wmma32(a, b, c);
#pragma unroll
    for (int r = 0; r < 8; ++r) {
      float xg = c[r];
      c[r] = 0.5f * xg * (1.0f + erff(xg * 0.70710678118f));  // exact GELU
    }
    store_c_f16(L.u.hbuf + nt * 16, 128, c, lane);
  }
  wave_sync();
#pragma unroll
  for (int nt = 0; nt < 2; ++nt) {
    v8f c = frag_c_bias(projB + nt * 16, lane);
#pragma unroll
    for (int kt = 0; kt < 4; ++kt) {
      v16h a2 = frag_a_f16(L.u.hbuf + kt * 32, 128, lane);
      v16h b  = frag_b_f16(projW + nt * 16 * 128 + kt * 32, 128, lane);
      c = wmma32(a2, b, c);
    }
    store_c_add(L.x + nt * 16, EMB, c, lane);
  }
}

// ---------------- kernels ----------------

struct K1Args {
  const int*   x_pep;
  const float* cmhc;
  const float* emb;
  const float* ln_g;  const float* ln_b;
  const float* pre_b; const float* sa_inb; const float* sa_ob;
  const float* cpre_b; const float* ca_inb; const float* ca_ob;
  const float* fc_b;  const float* proj_b;
  const float* h1_w1; const float* h1_b1;
  const half_t* w16;
  float* t_out;       // [B][32]
  float* a1_out;      // [B][128]
};

__global__ void __launch_bounds__(64)
transformer_kernel(K1Args args) {
  __shared__ WaveLds lds[2];
  const int wave = threadIdx.x >> 5;
  const int lane = threadIdx.x & 31;
  WaveLds& L = lds[wave];
  const int elem = blockIdx.x * 2 + wave;

  // phase 0: kick off this sample's cmhc -> LDS copy.
  // Async DMA path (no VGPR round-trip, tracked by ASYNCcnt); it overlaps
  // embedding + LN + the whole self-attention block of layer 0.
  {
    const float* src = args.cmhc + (size_t)elem * (MHC * EMB);
#if defined(HAVE_ASYNC_LDS)
    gchar_c* g = (gchar_c*)(const void*)src;
    lchar*   d = (lchar*)(void*)L.cmhc;
    for (int i = lane; i < (MHC * EMB) / 4; i += 32)  // 592 x b128
      __builtin_amdgcn_global_load_async_to_lds_b128(
          (gvec4*)(g + (size_t)i * 16), (lvec4*)(d + (size_t)i * 16), 0, 0);
#else
    const float4* s4 = reinterpret_cast<const float4*>(src);
    float4* d4 = reinterpret_cast<float4*>(L.cmhc);
    for (int i = lane; i < (MHC * EMB) / 4; i += 32) d4[i] = s4[i];
#endif
#pragma unroll
    for (int r = MHC; r < MHC_PAD; ++r) L.cmhc[r * EMB + lane] = 0.0f;
  }

  // phase 1: embedding + positional encoding + tanh  (lane = channel)
  {
    const int* tok = args.x_pep + (size_t)elem * SEQ;
    const float dscale = -0.28782313662425572f;  // -ln(10000)/32
    float div = __expf((float)(lane & ~1) * dscale);
    for (int row = 0; row < SEQ; ++row) {
      float pe = (lane & 1) ? __cosf(row * div) : __sinf(row * div);
      L.x[row * EMB + lane] = tanhf(args.emb[tok[row] * EMB + lane] + pe);
    }
    L.x[SEQ * EMB + lane] = 0.0f;
  }
  wave_sync();

  for (int l = 0; l < 3; ++l) {
    const half_t* preW  = args.w16 + l * 3 * 1024;
    const half_t* saW   = args.w16 +  9216 + l * 3072;
    const half_t* saOW  = args.w16 + 18432 + l * 1024;
    const half_t* cpreW = args.w16 + 21504 + l * 3072;
    const half_t* caW   = args.w16 + 30720 + l * 3072;
    const half_t* caOW  = args.w16 + 39936 + l * 1024;
    const half_t* fcW   = args.w16 + 43008 + l * 4096;
    const half_t* projW = args.w16 + 55296 + l * 4096;
    const float* lng   = args.ln_g   + l * 3 * 32;
    const float* lnb   = args.ln_b   + l * 3 * 32;
    const float* preB  = args.pre_b  + l * 3 * 32;
    const float* saB   = args.sa_inb + l * 96;
    const float* saOB  = args.sa_ob  + l * 32;
    const float* cpreB = args.cpre_b + l * 3 * 32;
    const float* caB   = args.ca_inb + l * 96;
    const float* caOB  = args.ca_ob  + l * 32;
    const float* fcB   = args.fc_b   + l * 128;
    const float* projB = args.proj_b + l * 32;

    // ---- self attention ----
    layer_norm(L.x, L.xn, lng, lnb, lane);
    wave_sync();
    proj32_f32(L.u.a.tmp, L.xn, preW, preB, lane);                  // q pre-proj
    wave_sync();
    proj32_f32(L.u.a.q, L.u.a.tmp, saW, saB, lane);                 // qh
    wave_sync();
    proj32_f32(L.u.a.tmp, L.xn, preW + 1024, preB + 32, lane);      // k pre-proj
    wave_sync();
    proj32_f16(L.khf, L.u.a.tmp, saW + 32 * 32, saB + 32, lane);    // kh
    wave_sync();
    proj32_f32(L.u.a.tmp, L.xn, preW + 2048, preB + 64, lane);      // v pre-proj
    wave_sync();
    proj32_f16(L.vhf, L.u.a.tmp, saW + 64 * 32, saB + 64, lane);    // vh
    wave_sync();
    attention(L.u.a.q, L.khf, L.vhf, SEQ, L.u.a.tmp, lane);
    wave_sync();
    proj32_add(L.x, L.u.a.tmp, saOW, saOB, lane);                   // residual
    wave_sync();

    // ---- cross attention ----
    layer_norm(L.x, L.xn, lng + 32, lnb + 32, lane);
    wave_sync();
    proj32_f32(L.u.a.tmp, L.xn, cpreW, cpreB, lane);                // cq pre-proj
    wave_sync();
    proj32_f32(L.u.a.q, L.u.a.tmp, caW, caB, lane);                 // qh
    wave_sync();
    if (l == 0) async_copy_wait();  // cmhc DMA must have landed before first use
    cross_kv(L.khf, L.cmhc, cpreW + 1024, cpreB + 32, caW + 32 * 32, caB + 32, lane);
    wave_sync();
    cross_kv(L.vhf, L.cmhc, cpreW + 2048, cpreB + 64, caW + 64 * 32, caB + 64, lane);
    wave_sync();
    attention(L.u.a.q, L.khf, L.vhf, MHC, L.u.a.tmp, lane);
    wave_sync();
    proj32_add(L.x, L.u.a.tmp, caOW, caOB, lane);
    wave_sync();

    // ---- FFN ----
    layer_norm(L.x, L.xn, lng + 64, lnb + 64, lane);
    wave_sync();
    ffn(L, fcW, fcB, projW, projB, lane);
    wave_sync();
  }

  // head stage 1: t = x[0,:], a1 = relu(t @ h1_w1^T + h1_b1)
  args.t_out[(size_t)elem * 32 + lane] = L.x[lane];
  float* a1 = args.a1_out + (size_t)elem * 128;
#pragma unroll
  for (int i = 0; i < 4; ++i) {
    int oc = lane + 32 * i;
    float s = args.h1_b1[oc];
    for (int k = 0; k < 32; ++k) s += L.x[k] * args.h1_w1[oc * 32 + k];
    a1[oc] = fmaxf(s, 0.0f);
  }
}

// f32 -> f16 weight repack (single pass over all matrix weights)
__global__ void convert_weights(const float* preW, const float* saW, const float* saOW,
                                const float* cpreW, const float* caW, const float* caOW,
                                const float* fcW, const float* projW, half_t* dst) {
  int i = blockIdx.x * blockDim.x + threadIdx.x;
  if (i >= W16_TOTAL) return;
  int off = i;
  float v;
  if (off < 9216)               v = preW[off];
  else if ((off -= 9216) < 9216) v = saW[off];
  else if ((off -= 9216) < 3072) v = saOW[off];
  else if ((off -= 3072) < 9216) v = cpreW[off];
  else if ((off -= 9216) < 9216) v = caW[off];
  else if ((off -= 9216) < 3072) v = caOW[off];
  else if ((off -= 3072) < 12288) v = fcW[off];
  else { off -= 12288;           v = projW[off]; }
  dst[i] = (_Float16)v;
}

// per-channel batch mean/var (one block per channel)
__global__ void col_reduce(const float* data, int rows, int cols, float* mean, float* var) {
  __shared__ float ssum[256], ssq[256];
  int ch = blockIdx.x;
  float s = 0.0f, q = 0.0f;
  for (int r = threadIdx.x; r < rows; r += blockDim.x) {
    float v = data[(size_t)r * cols + ch];
    s += v; q += v * v;
  }
  ssum[threadIdx.x] = s; ssq[threadIdx.x] = q;
  __syncthreads();
  for (int off = 128; off; off >>= 1) {
    if (threadIdx.x < off) {
      ssum[threadIdx.x] += ssum[threadIdx.x + off];
      ssq[threadIdx.x]  += ssq[threadIdx.x + off];
    }
    __syncthreads();
  }
  if (threadIdx.x == 0) {
    float m = ssum[0] / rows;
    mean[ch] = m;
    var[ch]  = ssq[0] / rows - m * m;
  }
}

// s = t + relu( BN(a1) @ h1_w2^T + h1_b2 )
__global__ void head_mid(const float* t, const float* a1, const float* mean1, const float* var1,
                         const float* bn_g, const float* bn_b, const float* w2, const float* b2,
                         float* s_out, int B) {
  int gid  = blockIdx.x * (blockDim.x >> 5) + (threadIdx.x >> 5);
  int lane = threadIdx.x & 31;
  if (gid >= B) return;
  const float* a = a1 + (size_t)gid * 128;
  float acc = b2[lane];
  for (int k = 0; k < 128; ++k) {
    float hk = (a[k] - mean1[k]) * rsqrtf(var1[k] + 1e-5f) * bn_g[k] + bn_b[k];
    acc += hk * w2[lane * 128 + k];
  }
  s_out[(size_t)gid * 32 + lane] = t[(size_t)gid * 32 + lane] + fmaxf(acc, 0.0f);
}

// r = relu( BN(s) @ h2_w1^T + h2_b1 )
__global__ void head_r(const float* s, const float* mean2, const float* var2,
                       const float* g2, const float* b2bn,
                       const float* w1, const float* b1, float* r, int B) {
  int idx = blockIdx.x * blockDim.x + threadIdx.x;
  if (idx >= B * 8) return;
  int e = idx >> 3, oc = idx & 7;
  const float* se = s + (size_t)e * 32;
  float acc = b1[oc];
  for (int k = 0; k < 32; ++k) {
    float v = (se[k] - mean2[k]) * rsqrtf(var2[k] + 1e-5f) * g2[k] + b2bn[k];
    acc += v * w1[oc * 32 + k];
  }
  r[(size_t)e * 8 + oc] = fmaxf(acc, 0.0f);
}

// out = sigmoid( BN(r) @ h2_w2^T + h2_b2 )
__global__ void head_out(const float* r, const float* mean3, const float* var3,
                         const float* g3, const float* b3,
                         const float* w2, const float* b2, float* out, int B) {
  int e = blockIdx.x * blockDim.x + threadIdx.x;
  if (e >= B) return;
  float acc = b2[0];
#pragma unroll
  for (int k = 0; k < 8; ++k) {
    float v = (r[(size_t)e * 8 + k] - mean3[k]) * rsqrtf(var3[k] + 1e-5f) * g3[k] + b3[k];
    acc += v * w2[k];
  }
  out[e] = 1.0f / (1.0f + __expf(-acc));
}

// ---------------- host ----------------

extern "C" void kernel_launch(void* const* d_in, const int* in_sizes, int n_in,
                              void* d_out, int out_size, void* d_ws, size_t ws_size,
                              hipStream_t stream) {
  const int B = NB;
  float* ws     = (float*)d_ws;
  float* t_buf  = ws;                                  // B*32
  float* a1_buf = t_buf + (size_t)B * 32;              // B*128
  float* s_buf  = a1_buf + (size_t)B * 128;            // B*32
  float* r_buf  = s_buf + (size_t)B * 32;              // B*8
  float* m1 = r_buf + (size_t)B * 8;  float* v1 = m1 + 128;
  float* m2 = v1 + 128;               float* v2 = m2 + 32;
  float* m3 = v2 + 32;                float* v3 = m3 + 8;
  half_t* w16 = (half_t*)(v3 + 8);                     // 67584 halfs

  convert_weights<<<(W16_TOTAL + 255) / 256, 256, 0, stream>>>(
      (const float*)d_in[5],  (const float*)d_in[7],  (const float*)d_in[9],
      (const float*)d_in[11], (const float*)d_in[13], (const float*)d_in[15],
      (const float*)d_in[17], (const float*)d_in[19], w16);

  K1Args args;
  args.x_pep  = (const int*)d_in[0];
  args.cmhc   = (const float*)d_in[1];
  args.emb    = (const float*)d_in[2];
  args.ln_g   = (const float*)d_in[3];
  args.ln_b   = (const float*)d_in[4];
  args.pre_b  = (const float*)d_in[6];
  args.sa_inb = (const float*)d_in[8];
  args.sa_ob  = (const float*)d_in[10];
  args.cpre_b = (const float*)d_in[12];
  args.ca_inb = (const float*)d_in[14];
  args.ca_ob  = (const float*)d_in[16];
  args.fc_b   = (const float*)d_in[18];
  args.proj_b = (const float*)d_in[20];
  args.h1_w1  = (const float*)d_in[21];
  args.h1_b1  = (const float*)d_in[22];
  args.w16    = w16;
  args.t_out  = t_buf;
  args.a1_out = a1_buf;

  transformer_kernel<<<B / 2, 64, 0, stream>>>(args);

  col_reduce<<<128, 256, 0, stream>>>(a1_buf, B, 128, m1, v1);
  head_mid<<<B / 8, 256, 0, stream>>>(t_buf, a1_buf, m1, v1,
                                      (const float*)d_in[23], (const float*)d_in[24],
                                      (const float*)d_in[25], (const float*)d_in[26],
                                      s_buf, B);
  col_reduce<<<32, 256, 0, stream>>>(s_buf, B, 32, m2, v2);
  head_r<<<(B * 8 + 255) / 256, 256, 0, stream>>>(s_buf, m2, v2,
                                                  (const float*)d_in[27], (const float*)d_in[28],
                                                  (const float*)d_in[29], (const float*)d_in[30],
                                                  r_buf, B);
  col_reduce<<<8, 256, 0, stream>>>(r_buf, B, 8, m3, v3);
  head_out<<<(B + 255) / 256, 256, 0, stream>>>(r_buf, m3, v3,
                                                (const float*)d_in[31], (const float*)d_in[32],
                                                (const float*)d_in[33], (const float*)d_in[34],
                                                (float*)d_out, B);
}
